// MultiHeadAttentionModel_51616916963873
// MI455X (gfx1250) — matrix operand order repaired and checked by reading, not compiled, not run on hardware
//
#include <hip/hip_runtime.h>
#include <stdint.h>

// Problem constants (match reference)
constexpr int B    = 16;
constexpr int P    = 1024;
constexpr int DIN  = 768;
constexpr int DK   = 768;
constexpr int H    = 12;
constexpr int HD   = DK / H;     // 64
constexpr int NCLS = 1000;
constexpr int N3   = 3 * DK;     // 2304

typedef __attribute__((ext_vector_type(16))) __bf16 v16bf;
typedef __attribute__((ext_vector_type(4)))  __bf16 v4bf;
typedef __attribute__((ext_vector_type(2)))  __bf16 v2bf;
typedef __attribute__((ext_vector_type(8)))  float  v8f;

__device__ __forceinline__ v8f wmma_bf16(v16bf a, v16bf b, v8f c) {
  return __builtin_amdgcn_wmma_f32_16x16x32_bf16(false, a, false, b, (short)0, c,
                                                 false, false);
}

// A-matrix (16x32, 16-bit) pair index: VGPR j holds two consecutive K
// values; lanes 16-31 take K 16..31 (contiguous, matches B loads).
__device__ __forceinline__ int a_kpair(int j, int hi) {
  return 2 * j + hi * 16;
}

// Gather one 16x32 bf16 A tile starting at `base` (row already selected).
__device__ __forceinline__ v16bf gather_a_bf16(const __bf16* base, int hi) {
  v16bf a;
#pragma unroll
  for (int j = 0; j < 8; ++j) {
    const v2bf p = *(const v2bf*)(base + a_kpair(j, hi));
    a[2 * j]     = p[0];
    a[2 * j + 1] = p[1];
  }
  return a;
}

// ---------------------------------------------------------------------------
// Kernel 0: zero the pooled-sum accumulator
// ---------------------------------------------------------------------------
__global__ void zero_kernel(float* __restrict__ p, int n) {
  int i = blockIdx.x * blockDim.x + threadIdx.x;
  if (i < n) p[i] = 0.0f;
}

// ---------------------------------------------------------------------------
// Prep A: x (fp32) -> xbf (bf16), vectorized 4-wide
// ---------------------------------------------------------------------------
__global__ void cvt_x_kernel(const float* __restrict__ x,
                             __bf16* __restrict__ xbf, int n4) {
  int i = blockIdx.x * blockDim.x + threadIdx.x;
  if (i >= n4) return;
  float4 f = ((const float4*)x)[i];
  v4bf o = {(__bf16)f.x, (__bf16)f.y, (__bf16)f.z, (__bf16)f.w};
  ((v4bf*)xbf)[i] = o;
}

// ---------------------------------------------------------------------------
// Prep B: Wqkv [DIN x N3] fp32 -> Wt [N3 x DIN] bf16 (LDS-tiled transpose)
// ---------------------------------------------------------------------------
__global__ __launch_bounds__(256)
void prep_w_kernel(const float* __restrict__ Wqkv, __bf16* __restrict__ Wt) {
  __shared__ __bf16 t[32][33];
  const int kt = blockIdx.y * 32, nt = blockIdx.x * 32;
  const int tx = threadIdx.x & 31, ty = threadIdx.x >> 5;
#pragma unroll
  for (int r = 0; r < 4; ++r)
    t[ty + 8 * r][tx] = (__bf16)Wqkv[(size_t)(kt + ty + 8 * r) * N3 + nt + tx];
  __syncthreads();
#pragma unroll
  for (int r = 0; r < 4; ++r)
    Wt[(size_t)(nt + ty + 8 * r) * DIN + kt + tx] = t[tx][ty + 8 * r];
}

// ---------------------------------------------------------------------------
// Kernel 1: QKV = x @ Wqkv (+pos). One wave per 32x64 tile: 2 A-tiles x
// 4 B-tiles -> 8 WMMAs per 32-k step. Two-stage ping-pong buffers (loads
// refill in place after consumption -> no register rotation copies).
// Q,K stored row-major bf16; V stored TRANSPOSED [B, DK, P].
// ---------------------------------------------------------------------------
__global__ __launch_bounds__(32)
void qkv_kernel(const __bf16* __restrict__ xbf, const __bf16* __restrict__ Wt,
                const float* __restrict__ pos,
                __bf16* __restrict__ Qw, __bf16* __restrict__ Kw,
                __bf16* __restrict__ Vt) {
  const int n0   = blockIdx.x * 64;
  const int m0   = blockIdx.y * 32;
  const int lane = threadIdx.x & 31;
  const int mn   = lane & 15;
  const int hi   = lane >> 4;

  const __bf16* ar0  = xbf + (size_t)(m0 + mn) * DIN;
  const __bf16* ar1  = ar0 + (size_t)16 * DIN;
  const __bf16* brow = Wt + (size_t)(n0 + mn) * DIN + hi * 16;

  v8f acc0[4] = {}, acc1[4] = {};

  // Stage A holds k-chunk k0, stage B holds k0+32.
  v16bf aA0 = gather_a_bf16(ar0, hi);
  v16bf aA1 = gather_a_bf16(ar1, hi);
  v16bf bA[4];
#pragma unroll
  for (int t = 0; t < 4; ++t)
    bA[t] = *(const v16bf*)(brow + (size_t)t * 16 * DIN);
  v16bf aB0 = gather_a_bf16(ar0 + 32, hi);
  v16bf aB1 = gather_a_bf16(ar1 + 32, hi);
  v16bf bB[4];
#pragma unroll
  for (int t = 0; t < 4; ++t)
    bB[t] = *(const v16bf*)(brow + (size_t)t * 16 * DIN + 32);

  for (int k0 = 64; k0 < DIN; k0 += 64) {
    // consume stage A (k0-64), refill with k0
#pragma unroll
    for (int t = 0; t < 4; ++t) {
      acc0[t] = wmma_bf16(aA0, bA[t], acc0[t]);
      acc1[t] = wmma_bf16(aA1, bA[t], acc1[t]);
    }
    aA0 = gather_a_bf16(ar0 + k0, hi);
    aA1 = gather_a_bf16(ar1 + k0, hi);
#pragma unroll
    for (int t = 0; t < 4; ++t)
      bA[t] = *(const v16bf*)(brow + (size_t)t * 16 * DIN + k0);
    // consume stage B (k0-32), refill with k0+32
#pragma unroll
    for (int t = 0; t < 4; ++t) {
      acc0[t] = wmma_bf16(aB0, bB[t], acc0[t]);
      acc1[t] = wmma_bf16(aB1, bB[t], acc1[t]);
    }
    aB0 = gather_a_bf16(ar0 + k0 + 32, hi);
    aB1 = gather_a_bf16(ar1 + k0 + 32, hi);
#pragma unroll
    for (int t = 0; t < 4; ++t)
      bB[t] = *(const v16bf*)(brow + (size_t)t * 16 * DIN + k0 + 32);
  }
#pragma unroll
  for (int t = 0; t < 4; ++t) {                  // drain both stages
    acc0[t] = wmma_bf16(aA0, bA[t], acc0[t]);
    acc1[t] = wmma_bf16(aA1, bA[t], acc1[t]);
  }
#pragma unroll
  for (int t = 0; t < 4; ++t) {
    acc0[t] = wmma_bf16(aB0, bB[t], acc0[t]);
    acc1[t] = wmma_bf16(aB1, bB[t], acc1[t]);
  }

#pragma unroll
  for (int t = 0; t < 4; ++t) {
    const int col = n0 + t * 16 + mn;
    const int seg = col / DK;                     // 0:Q 1:K 2:V
    const int dc  = col - seg * DK;
#pragma unroll
    for (int mi = 0; mi < 2; ++mi) {
#pragma unroll
      for (int r = 0; r < 8; ++r) {
        const int m = m0 + mi * 16 + r + 8 * hi;  // global row = b*P + p
        const int p = m & (P - 1);
        const int bb = m >> 10;
        const float v =
            (mi ? acc1[t][r] : acc0[t][r]) + pos[(size_t)p * DK + dc];
        if (seg == 0)      Qw[(size_t)m * DK + dc] = (__bf16)v;
        else if (seg == 1) Kw[(size_t)m * DK + dc] = (__bf16)v;
        else               Vt[(size_t)(bb * DK + dc) * P + p] = (__bf16)v;
      }
    }
  }
}

// ---------------------------------------------------------------------------
// Kernel 2: fused scores -> softmax -> (weights @ V) -> pooled column sums.
// Block = 256 threads (8 waves); 16 query rows of one (b,h).
// LDS: 66 KB fp32 score strip + 33 KB bf16 weight strip + 2 KB Q strip.
// ---------------------------------------------------------------------------
constexpr int SLD = P + 8;                        // padded fp32 row stride
constexpr int WLD = P + 8;                        // padded bf16 row stride

__global__ __launch_bounds__(256)
void attn_kernel(const __bf16* __restrict__ Qw, const __bf16* __restrict__ Kw,
                 const __bf16* __restrict__ Vt,
                 float* __restrict__ attn_out,    // d_out + 16000
                 float* __restrict__ pooled_sum) {
  __shared__ float  sc[16 * SLD];                 // raw scores (fp32)
  __shared__ __bf16 wlds[16 * WLD];               // normalized weights (bf16)
  __shared__ __bf16 qlds[16 * HD];                // Q strip

  const int bh = blockIdx.x;                      // b*H + h
  const int b  = bh / H, h = bh % H;
  const int q0 = blockIdx.y * 16;
  const int tid  = threadIdx.x;
  const int wave = tid >> 5;
  const int lane = tid & 31;
  const int mn   = lane & 15;
  const int hi   = lane >> 4;

  const __bf16* Qb  = Qw + ((size_t)(b * P + q0) * DK + h * HD);
  const __bf16* Kb  = Kw + ((size_t)b * P * DK + h * HD);
  const __bf16* Vtb = Vt + (size_t)(b * DK + h * HD) * P;

  // ---- Stage Q strip [16 x 64] bf16 into LDS with async copy (ASYNCcnt) --
  {
    const int row = tid >> 4, ch = tid & 15;      // 8 bytes per thread
    const uint64_t ga = (uint64_t)(uintptr_t)(Qb + (size_t)row * DK + ch * 4);
    const uint32_t la = (uint32_t)(uintptr_t)(&qlds[row * HD + ch * 4]);
    asm volatile("global_load_async_to_lds_b64 %0, %1, off"
                 :: "v"(la), "v"(ga) : "memory");
    asm volatile("s_wait_asynccnt 0x0" ::: "memory");
  }
  __syncthreads();

  // ---- Phase 1: scores[16 x 1024] = Q_strip @ K^T * 1/sqrt(HD) ------------
  {
    // A tiles are invariant across column tiles: hoist both k-halves.
    const v16bf a0 = gather_a_bf16(&qlds[mn * HD], hi);
    const v16bf a1 = gather_a_bf16(&qlds[mn * HD + 32], hi);
    int col0 = wave * 128;
    v16bf b0 = *(const v16bf*)(Kb + (size_t)(col0 + mn) * DK + hi * 16);
    v16bf b1 = *(const v16bf*)(Kb + (size_t)(col0 + mn) * DK + 32 + hi * 16);
#pragma unroll
    for (int nt = 0; nt < 8; ++nt) {
      v16bf nb0 = b0, nb1 = b1;
      if (nt < 7) {                               // next column tile in flight
        const size_t roff = (size_t)(col0 + 16 + mn) * DK;
        nb0 = *(const v16bf*)(Kb + roff + hi * 16);
        nb1 = *(const v16bf*)(Kb + roff + 32 + hi * 16);
      }
      v8f c = {};
      c = wmma_bf16(a0, b0, c);
      c = wmma_bf16(a1, b1, c);
#pragma unroll
      for (int r = 0; r < 8; ++r)
        sc[(r + 8 * hi) * SLD + col0 + mn] = c[r] * 0.125f;  // 1/sqrt(64)
      col0 += 16;
      b0 = nb0;
      b1 = nb1;
    }
  }
  __syncthreads();

  // ---- Phase 2: row softmax (16 threads/row); write fp32 out + bf16 LDS --
  {
    const int row = tid >> 4;
    const int j0  = tid & 15;
    float mx = -1e30f;
    for (int j = j0; j < P; j += 16) mx = fmaxf(mx, sc[row * SLD + j]);
#pragma unroll
    for (int m = 1; m < 16; m <<= 1) mx = fmaxf(mx, __shfl_xor(mx, m, 16));
    float sum = 0.0f;
    for (int j = j0; j < P; j += 16) sum += __expf(sc[row * SLD + j] - mx);
#pragma unroll
    for (int m = 1; m < 16; m <<= 1) sum += __shfl_xor(sum, m, 16);
    const float inv = 1.0f / sum;
    float* orow = attn_out + ((size_t)bh * P + q0 + row) * (size_t)P;
    for (int j = j0; j < P; j += 16) {
      const float w = __expf(sc[row * SLD + j] - mx) * inv;
      orow[j] = w;                                // mandatory fp32 output
      wlds[row * WLD + j] = (__bf16)w;            // bf16 copy for AV GEMM
    }
  }
  __syncthreads();

  // ---- Phase 3: out_tile[16 x 64] = weights @ V_head; pool over rows -----
  // Two-stage ping-pong over the 32 k-chunks (no rotation copies).
  if (wave < 4) {
    const int n0 = wave * 16;
    const __bf16* vrow = Vtb + (size_t)(n0 + mn) * P + hi * 16;
    const __bf16* wrow = &wlds[mn * WLD];

    v16bf a0 = gather_a_bf16(wrow, hi);
    v16bf b0 = *(const v16bf*)(vrow);
    v16bf a1 = gather_a_bf16(wrow + 32, hi);
    v16bf b1 = *(const v16bf*)(vrow + 32);
    v8f c = {};
    for (int kk = 2; kk < 32; kk += 2) {
      c = wmma_bf16(a0, b0, c);
      a0 = gather_a_bf16(wrow + kk * 32, hi);
      b0 = *(const v16bf*)(vrow + kk * 32);
      c = wmma_bf16(a1, b1, c);
      a1 = gather_a_bf16(wrow + (kk + 1) * 32, hi);
      b1 = *(const v16bf*)(vrow + (kk + 1) * 32);
    }
    c = wmma_bf16(a0, b0, c);
    c = wmma_bf16(a1, b1, c);

    float s = 0.0f;
#pragma unroll
    for (int r = 0; r < 8; ++r) s += c[r];
    s += __shfl_xor(s, 16, 32);                   // add other half-wave
    if (hi == 0)
      atomicAdd(&pooled_sum[b * DK + h * HD + n0 + mn], s);
  }
}

// ---------------------------------------------------------------------------
// Kernel 3a: proj = (pooled_sum / P) @ Wo + bo          [16 x 768]
// ---------------------------------------------------------------------------
__global__ void proj_kernel(const float* __restrict__ pooled_sum,
                            const float* __restrict__ Wo,
                            const float* __restrict__ bo,
                            float* __restrict__ proj) {
  const int i = blockIdx.x * blockDim.x + threadIdx.x;
  if (i >= B * DK) return;
  const int bb = i / DK, j = i % DK;
  const float invP = 1.0f / (float)P;
  float acc = bo[j];
  for (int d = 0; d < DK; ++d)
    acc += (pooled_sum[bb * DK + d] * invP) * Wo[(size_t)d * DK + j];
  proj[i] = acc;
}

// ---------------------------------------------------------------------------
// Kernel 3b: logits = proj @ Wc + bc                    [16 x 1000]
// ---------------------------------------------------------------------------
__global__ void cls_kernel(const float* __restrict__ proj,
                           const float* __restrict__ Wc,
                           const float* __restrict__ bc,
                           float* __restrict__ logits) {
  const int i = blockIdx.x * blockDim.x + threadIdx.x;
  if (i >= B * NCLS) return;
  const int bb = i / NCLS, cidx = i % NCLS;
  float acc = bc[cidx];
  for (int j = 0; j < DK; ++j)
    acc += proj[bb * DK + j] * Wc[(size_t)j * NCLS + cidx];
  logits[i] = acc;
}

// ---------------------------------------------------------------------------
extern "C" void kernel_launch(void* const* d_in, const int* in_sizes, int n_in,
                              void* d_out, int out_size, void* d_ws, size_t ws_size,
                              hipStream_t stream) {
  const float* x    = (const float*)d_in[0];
  const float* Wqkv = (const float*)d_in[1];
  const float* pos  = (const float*)d_in[2];
  const float* Wo   = (const float*)d_in[3];
  const float* bo   = (const float*)d_in[4];
  const float* Wc   = (const float*)d_in[5];
  const float* bc   = (const float*)d_in[6];

  float* logits = (float*)d_out;                       // [16 x 1000]
  float* attnW  = (float*)d_out + (size_t)B * NCLS;    // [16,12,1024,1024]

  // Workspace layout (~104 MB):
  const size_t xe  = (size_t)B * P * DIN;              // 12,582,912
  const size_t we  = (size_t)N3 * DIN;                 //  1,769,472
  const size_t qe  = (size_t)B * P * DK;               // 12,582,912
  __bf16* xbf = (__bf16*)d_ws;
  __bf16* Wt  = xbf + xe;
  __bf16* Qw  = Wt + we;
  __bf16* Kw  = Qw + qe;
  __bf16* Vt  = Kw + qe;                               // transposed [B, DK, P]
  float* pooled = (float*)(Vt + qe);                   // [B, DK]
  float* proj   = pooled + B * DK;                     // [B, DK]

  zero_kernel<<<dim3((B * DK + 255) / 256), 256, 0, stream>>>(pooled, B * DK);

  cvt_x_kernel<<<dim3((int)(xe / 4 + 255) / 256), 256, 0, stream>>>(
      x, xbf, (int)(xe / 4));

  prep_w_kernel<<<dim3(N3 / 32, DIN / 32), 256, 0, stream>>>(Wqkv, Wt);

  qkv_kernel<<<dim3(N3 / 64, (B * P) / 32), 32, 0, stream>>>(
      xbf, Wt, pos, Qw, Kw, Vt);

  attn_kernel<<<dim3(B * H, P / 16), 256, 0, stream>>>(
      Qw, Kw, Vt, attnW, pooled);

  proj_kernel<<<dim3((B * DK + 255) / 256), 256, 0, stream>>>(
      pooled, Wo, bo, proj);

  cls_kernel<<<dim3((B * NCLS + 255) / 256), 256, 0, stream>>>(
      proj, Wc, bc, logits);
}
